// PairformerBlock_67259187855462
// MI455X (gfx1250) — matrix operand order repaired and checked by reading, not compiled, not run on hardware
//
#include <hip/hip_runtime.h>

// ---------------------------------------------------------------------------
// Pairformer block for MI455X (gfx1250, wave32, WMMA bf16 -> f32 accumulate)
//
// Reference shapes: L=256, C_Z=128, C_S=384, H=8, D_z=16, D_s=48.
// All matmuls use V_WMMA_F32_16X16X32_BF16 (f32 accumulation, bf16 operands).
// Triangle-einsum GEMM stages its bf16 tiles with GLOBAL_LOAD_ASYNC_TO_LDS
// (ASYNCcnt-tracked) when the toolchain exposes the gfx1250 builtin.
// ---------------------------------------------------------------------------

#define DEV __device__ __forceinline__

typedef __attribute__((ext_vector_type(16))) __bf16 bf16x16;
typedef __attribute__((ext_vector_type(8)))  float  f32x8;
typedef __attribute__((ext_vector_type(4)))  float  f32x4;

union FragU { bf16x16 v; f32x4 f[2]; };

#if defined(__HIP_DEVICE_COMPILE__) && \
    __has_builtin(__builtin_amdgcn_global_load_async_to_lds_b128)
#define HAS_ASYNC_LDS 1
#else
#define HAS_ASYNC_LDS 0
#endif

#if HAS_ASYNC_LDS
// Builtin signature (from hipcc diagnostic): param0 is
// 'int __attribute__((vector_size(16))) addrspace(1)*' — typed v4i pointers.
typedef int v4i_ __attribute__((vector_size(16)));
typedef __attribute__((address_space(1))) v4i_ g_v4i;
typedef __attribute__((address_space(3))) v4i_ l_v4i;
DEV void async_copy16(void* lds, const void* g) {
  __builtin_amdgcn_global_load_async_to_lds_b128((g_v4i*)g, (l_v4i*)lds, 0, 0);
}
DEV void wait_async0() {
#if __has_builtin(__builtin_amdgcn_s_wait_asynccnt)
  __builtin_amdgcn_s_wait_asynccnt(0);
#else
  asm volatile("s_wait_asynccnt 0" ::: "memory");
#endif
}
#endif

// A-matrix (16x32, MxK) fragment per CDNA5 layout:
//   lane: m = lane&15, half = lane>>4
//   VGPR 0..3 : K = half*8 + [0..7]   (16 bytes contiguous)
//   VGPR 4..7 : K = 16 + half*8 + [0..7]
DEV bf16x16 frag_a_ld(const __bf16* T, int stride, int row0, int lane) {
  int m = row0 + (lane & 15), hl = lane >> 4;
  const __bf16* p = T + (size_t)m * stride;
  FragU u;
  u.f[0] = *(const f32x4*)(p + hl * 8);
  u.f[1] = *(const f32x4*)(p + 16 + hl * 8);
  return u.v;
}

// B-matrix (32x16, KxN) fragment, operand tile stored transposed [n][k]:
//   lane: n = lane&15, half = lane>>4 ; elements cover K = half*8 + [0..7], 16 + half*8 + [0..7]
DEV bf16x16 frag_b_ld(const __bf16* T, int stride, int col0, int lane) {
  int n = col0 + (lane & 15), hl = lane >> 4;
  const __bf16* p = T + (size_t)n * stride;
  FragU u;
  u.f[0] = *(const f32x4*)(p + hl * 8);
  u.f[1] = *(const f32x4*)(p + 16 + hl * 8);
  return u.v;
}

DEV f32x8 wmma_bf16(bf16x16 a, bf16x16 b, f32x8 c) {
  return __builtin_amdgcn_wmma_f32_16x16x32_bf16(false, a, false, b,
                                                 (short)0, c, false, false);
}

DEV float sigmoidf_(float x) { return 1.f / (1.f + __expf(-x)); }

// ---------------------------------------------------------------------------
// LayerNorm: one block (128 thr) per row
// ---------------------------------------------------------------------------
__global__ void __launch_bounds__(128)
ln_kernel(const float* __restrict__ x, const float* __restrict__ g,
          const float* __restrict__ b, float* __restrict__ y, int C) {
  size_t row = blockIdx.x;
  const float* xr = x + row * C;
  float* yr = y + row * C;
  float s = 0.f, s2 = 0.f;
  for (int c = threadIdx.x; c < C; c += 128) {
    float v = xr[c];
    s += v; s2 += v * v;
  }
  __shared__ float rs[128], rq[128];
  rs[threadIdx.x] = s; rq[threadIdx.x] = s2;
  __syncthreads();
  for (int st = 64; st > 0; st >>= 1) {
    if ((int)threadIdx.x < st) {
      rs[threadIdx.x] += rs[threadIdx.x + st];
      rq[threadIdx.x] += rq[threadIdx.x + st];
    }
    __syncthreads();
  }
  float mean = rs[0] / C;
  float var  = rq[0] / C - mean * mean;
  float inv  = rsqrtf(var + 1e-5f);
  for (int c = threadIdx.x; c < C; c += 128)
    yr[c] = (xr[c] - mean) * inv * g[c] + b[c];
}

// ---------------------------------------------------------------------------
// Generic WMMA GEMM: Y[M,N] = act(A[M,K] @ W[K,N] + bias)
// Block: 128 thr (4 waves), tile 64x64, K-step 32. A converted to bf16 in LDS,
// W staged transposed ([n][k]) in LDS. K must be a multiple of 32.
// Next K-tile is prefetched (global_prefetch_b8) while the current one runs.
// ---------------------------------------------------------------------------
template <typename AT, typename OT>
__global__ void __launch_bounds__(128)
gemm_kernel(const AT* __restrict__ A, const float* __restrict__ W,
            const float* __restrict__ bias, OT* __restrict__ Y,
            int M, int N, int K, int relu) {
  __shared__ __align__(16) __bf16 As[64][32];
  __shared__ __align__(16) __bf16 Bs[64][32];
  int n0 = blockIdx.x * 64, m0 = blockIdx.y * 64;
  int wave = threadIdx.x >> 5, lane = threadIdx.x & 31;

  f32x8 acc[4] = {f32x8{}, f32x8{}, f32x8{}, f32x8{}};

  for (int k0 = 0; k0 < K; k0 += 32) {
    if (k0 + 32 < K) {
      // prefetch next A / W tile rows into cache while this tile computes
      int pr = threadIdx.x >> 1;
      int m = m0 + pr;
      if ((threadIdx.x & 1) == 0) {
        if (m < M) __builtin_prefetch(&A[(size_t)m * K + (k0 + 32)], 0, 1);
      } else {
        __builtin_prefetch(&W[(size_t)(k0 + 32 + (pr & 31)) * N + n0], 0, 1);
      }
    }
    for (int t = threadIdx.x; t < 64 * 32; t += 128) {
      int r = t >> 5, kk = t & 31;
      int m = m0 + r;
      float av = (m < M) ? (float)A[(size_t)m * K + (k0 + kk)] : 0.f;
      As[r][kk] = (__bf16)av;
      int n = n0 + r;
      float wv = (n < N) ? W[(size_t)(k0 + kk) * N + n] : 0.f;
      Bs[r][kk] = (__bf16)wv;
    }
    __syncthreads();
    bf16x16 a = frag_a_ld(&As[0][0], 32, wave * 16, lane);
#pragma unroll
    for (int nt = 0; nt < 4; ++nt) {
      bf16x16 b = frag_b_ld(&Bs[0][0], 32, nt * 16, lane);
      acc[nt] = wmma_bf16(a, b, acc[nt]);
    }
    __syncthreads();
  }

  int nl = lane & 15, hl = lane >> 4;
#pragma unroll
  for (int nt = 0; nt < 4; ++nt) {
    int n = n0 + nt * 16 + nl;
    if (n >= N) continue;
    float bv = bias ? bias[n] : 0.f;
#pragma unroll
    for (int r = 0; r < 8; ++r) {
      int m = m0 + wave * 16 + hl * 8 + r;
      if (m >= M) continue;
      float vv = acc[nt][r] + bv;
      if (relu) vv = fmaxf(vv, 0.f);
      Y[(size_t)m * N + n] = (OT)vv;
    }
  }
}

// ---------------------------------------------------------------------------
// Triangle einsum as 128 batched per-channel GEMMs:
//   out[i,j,c] = sum_k At[c][i][k] * Bt[c][j][k]
// Operands are pre-packed bf16, so tiles are staged with async LDS copies
// (GLOBAL_LOAD_ASYNC_TO_LDS_B128) when available: 2x 16B chunks per thread
// per tile, waited with s_wait_asynccnt + barrier.
// ---------------------------------------------------------------------------
__global__ void __launch_bounds__(128)
tri_gemm_kernel(const __bf16* __restrict__ At, const __bf16* __restrict__ Bt,
                float* __restrict__ out) {
  const int L = 256, C = 128;
  __shared__ __align__(16) __bf16 As[64][32];
  __shared__ __align__(16) __bf16 Bs[64][32];
  int j0 = blockIdx.x * 64, i0 = blockIdx.y * 64, c = blockIdx.z;
  int wave = threadIdx.x >> 5, lane = threadIdx.x & 31;
  const __bf16* Ac = At + (size_t)c * L * L;
  const __bf16* Bc = Bt + (size_t)c * L * L;

  f32x8 acc[4] = {f32x8{}, f32x8{}, f32x8{}, f32x8{}};

  for (int k0 = 0; k0 < L; k0 += 32) {
#if HAS_ASYNC_LDS
    // 64x32 bf16 tile = 256 x 16B chunks; 128 threads issue 2 chunks each
    // per operand, ASYNCcnt-tracked, no VGPR round trip.
    for (int ch = threadIdx.x; ch < 256; ch += 128) {
      int r = ch >> 2, seg = ch & 3;
      async_copy16(&As[r][seg * 8], Ac + (size_t)(i0 + r) * L + k0 + seg * 8);
      async_copy16(&Bs[r][seg * 8], Bc + (size_t)(j0 + r) * L + k0 + seg * 8);
    }
    wait_async0();
    __syncthreads();
#else
    for (int t = threadIdx.x; t < 64 * 32; t += 128) {
      int r = t >> 5, kk = t & 31;
      As[r][kk] = Ac[(size_t)(i0 + r) * L + (k0 + kk)];
      Bs[r][kk] = Bc[(size_t)(j0 + r) * L + (k0 + kk)];
    }
    __syncthreads();
#endif
    bf16x16 a = frag_a_ld(&As[0][0], 32, wave * 16, lane);
#pragma unroll
    for (int nt = 0; nt < 4; ++nt) {
      bf16x16 b = frag_b_ld(&Bs[0][0], 32, nt * 16, lane);
      acc[nt] = wmma_bf16(a, b, acc[nt]);
    }
    __syncthreads();
  }

  int nl = lane & 15, hl = lane >> 4;
#pragma unroll
  for (int nt = 0; nt < 4; ++nt) {
    int n = j0 + nt * 16 + nl;
#pragma unroll
    for (int r = 0; r < 8; ++r) {
      int m = i0 + wave * 16 + hl * 8 + r;
      out[((size_t)m * L + n) * C + c] = acc[nt][r];
    }
  }
}

// ---------------------------------------------------------------------------
// Packers / elementwise
// ---------------------------------------------------------------------------
// mode 0: out[c][i][k] = in[(i*L+k)*C + c]   ('ikc' operand)
// mode 1: out[c][i][k] = in[(k*L+i)*C + c]   ('kic' operand)
__global__ void pack_cik_kernel(const float* __restrict__ in,
                                __bf16* __restrict__ out, int mode) {
  const int L = 256, C = 128;
  size_t idx = (size_t)blockIdx.x * 256 + threadIdx.x;
  if (idx >= (size_t)C * L * L) return;
  int k = idx % L;
  size_t r = idx / L;
  int i = r % L;
  int c = r / L;
  float v = (mode == 0) ? in[((size_t)i * L + k) * C + c]
                        : in[((size_t)k * L + i) * C + c];
  out[idx] = (__bf16)v;
}

// out[((h*L+i)*L+j)*16+d] = in[(i*L+j)*128 + h*16 + d]
__global__ void pack_heads_kernel(const float* __restrict__ in,
                                  __bf16* __restrict__ out) {
  const int L = 256;
  size_t idx = (size_t)blockIdx.x * 256 + threadIdx.x;
  if (idx >= (size_t)8 * L * L * 16) return;
  int d = idx & 15;
  size_t r = idx >> 4;
  int j = r % L; r /= L;
  int i = r % L;
  int h = r / L;
  out[idx] = (__bf16)in[((size_t)i * L + j) * 128 + h * 16 + d];
}

__global__ void sigmul_kernel(const float* a, const float* g, float* y, size_t n) {
  size_t i = (size_t)blockIdx.x * 256 + threadIdx.x;
  if (i < n) y[i] = a[i] * sigmoidf_(g[i]);
}

__global__ void gate_residual_kernel(const float* base, const float* gl,
                                     const float* x, float* out, size_t n) {
  size_t i = (size_t)blockIdx.x * 256 + threadIdx.x;
  if (i < n) out[i] = base[i] + sigmoidf_(gl[i]) * x[i];
}

__global__ void add_kernel(float* y, const float* x, size_t n) {
  size_t i = (size_t)blockIdx.x * 256 + threadIdx.x;
  if (i < n) y[i] += x[i];
}

// out[(j*L+i)*C+c] = in[(i*L+j)*C+c]
__global__ void transpose_kernel(const float* __restrict__ in,
                                 float* __restrict__ out) {
  const int L = 256, C = 128;
  size_t idx = (size_t)blockIdx.x * 256 + threadIdx.x;
  if (idx >= (size_t)L * L * C) return;
  int c = idx % C;
  size_t r = idx / C;
  int j = r % L;
  int i = r / L;
  out[((size_t)j * L + i) * C + c] = in[idx];
}

// ---------------------------------------------------------------------------
// Triangle attention: one wave per (h, i, 16-row j-tile).
// S = Q K^T * 0.25 + bias[h,j,k] (D=16 zero-padded into K=32 fragments),
// softmax over k in LDS, then O = P V via WMMA (V tiles transposed in LDS).
// ---------------------------------------------------------------------------
__global__ void __launch_bounds__(32)
tri_attn_kernel(const __bf16* __restrict__ qp, const __bf16* __restrict__ kp,
                const __bf16* __restrict__ vp, const float* __restrict__ biasb,
                float* __restrict__ obuf) {
  const int L = 256;
  const int jt = blockIdx.x, i = blockIdx.y, h = blockIdx.z;
  const int lane = threadIdx.x;
  const int nl = lane & 15, hl = lane >> 4;

  __shared__ __align__(16) float  Ssm[16][256];
  __shared__ __align__(16) __bf16 Psm[16][256];
  __shared__ __align__(16) __bf16 Vsm[16][32];

  const f32x4 zero4 = {0.f, 0.f, 0.f, 0.f};
  const size_t head_base = ((size_t)h * L + i) * L;

  FragU qa;  // A fragment: rows j, K = d (upper 16 K zero)
  {
    const __bf16* qrow = qp + (head_base + (size_t)(jt * 16 + nl)) * 16;
    qa.f[0] = *(const f32x4*)(qrow + hl * 8);
    qa.f[1] = zero4;
  }

  for (int kt = 0; kt < 16; ++kt) {
    FragU kb;  // B fragment: cols k, K = d (upper 16 K zero)
    {
      const __bf16* krow = kp + (head_base + (size_t)(kt * 16 + nl)) * 16;
      kb.f[0] = *(const f32x4*)(krow + hl * 8);
      kb.f[1] = zero4;
    }
    f32x8 s = {};
    s = wmma_bf16(qa.v, kb.v, s);
#pragma unroll
    for (int r = 0; r < 8; ++r) {
      int m = hl * 8 + r;
      int jg = jt * 16 + m, kg = kt * 16 + nl;
      Ssm[m][kg] = s[r] * 0.25f + biasb[((size_t)jg * L + kg) * 8 + h];
    }
  }
  __syncthreads();

  if (lane < 16) {
    float mx = -3.0e38f;
    for (int k2 = 0; k2 < 256; ++k2) mx = fmaxf(mx, Ssm[lane][k2]);
    float sum = 0.f;
    for (int k2 = 0; k2 < 256; ++k2) {
      float e = __expf(Ssm[lane][k2] - mx);
      Ssm[lane][k2] = e;
      sum += e;
    }
    float inv = 1.f / sum;
    for (int k2 = 0; k2 < 256; ++k2)
      Psm[lane][k2] = (__bf16)(Ssm[lane][k2] * inv);
  }
  __syncthreads();

  f32x8 o = {};
  for (int kt = 0; kt < 8; ++kt) {  // 32 k-values per step
    for (int t = lane; t < 16 * 32; t += 32) {
      int d = t >> 5, kk = t & 31;
      Vsm[d][kk] = vp[(head_base + (size_t)(kt * 32 + kk)) * 16 + d];
    }
    __syncthreads();
    FragU pa;
    const __bf16* prow = &Psm[nl][0];
    pa.f[0] = *(const f32x4*)(prow + kt * 32 + hl * 8);
    pa.f[1] = *(const f32x4*)(prow + kt * 32 + 16 + hl * 8);
    bf16x16 vb = frag_b_ld(&Vsm[0][0], 32, 0, lane);
    o = wmma_bf16(pa.v, vb, o);
    __syncthreads();
  }

#pragma unroll
  for (int r = 0; r < 8; ++r) {
    int m = hl * 8 + r, jg = jt * 16 + m;
    obuf[((size_t)i * L + jg) * 128 + h * 16 + nl] = o[r];
  }
}

// ---------------------------------------------------------------------------
// Single-track attention with pair bias (small: H=8, L=256, D=48) — VALU.
// Block (64 thr) per (i, h).
// ---------------------------------------------------------------------------
__global__ void __launch_bounds__(64)
attn_single_kernel(const float* __restrict__ q, const float* __restrict__ k,
                   const float* __restrict__ v, const float* __restrict__ biasb,
                   float* __restrict__ o) {
  const int L = 256, CS = 384, D = 48;
  int i = blockIdx.x, h = blockIdx.y;
  int t = threadIdx.x;
  __shared__ float Arow[256];
  __shared__ float qs[48];
  __shared__ float red[64];

  if (t < D) qs[t] = q[(size_t)i * CS + h * D + t];
  __syncthreads();

  for (int j = t; j < L; j += 64) {
    float acc = 0.f;
    const float* kr = k + (size_t)j * CS + h * D;
    for (int d = 0; d < D; ++d) acc += qs[d] * kr[d];
    Arow[j] = acc * 0.14433756729740643f + biasb[((size_t)i * L + j) * 8 + h];
  }
  __syncthreads();

  float mx = -3.0e38f;
  for (int j = t; j < L; j += 64) mx = fmaxf(mx, Arow[j]);
  red[t] = mx; __syncthreads();
  for (int st = 32; st > 0; st >>= 1) {
    if (t < st) red[t] = fmaxf(red[t], red[t + st]);
    __syncthreads();
  }
  mx = red[0]; __syncthreads();

  float sum = 0.f;
  for (int j = t; j < L; j += 64) {
    float e = __expf(Arow[j] - mx);
    Arow[j] = e;
    sum += e;
  }
  red[t] = sum; __syncthreads();
  for (int st = 32; st > 0; st >>= 1) {
    if (t < st) red[t] += red[t + st];
    __syncthreads();
  }
  float inv = 1.f / red[0];
  __syncthreads();

  if (t < D) {
    float acc = 0.f;
    for (int j = 0; j < L; ++j) acc += Arow[j] * v[(size_t)j * CS + h * D + t];
    o[(size_t)i * CS + h * D + t] = acc * inv;
  }
}

// ---------------------------------------------------------------------------
// Host orchestration
// ---------------------------------------------------------------------------
extern "C" void kernel_launch(void* const* d_in, const int* in_sizes, int n_in,
                              void* d_out, int out_size, void* d_ws, size_t ws_size,
                              hipStream_t stream) {
  (void)in_sizes; (void)n_in; (void)out_size; (void)ws_size;
  const int L = 256, CZ = 128, CS = 384, H = 8;
  const size_t NP = (size_t)L * L;  // 65536 pair rows

  // --- input pointer map (setup_inputs() dict order, recursive insertion) ---
  // 0: single [L,CS]   1: pair [L,L,CZ]
  // tri_out  2..17 : ln.g ln.b lp.W lp.b rp.W rp.b lg.W lg.b rg.W rg.b og.W og.b op.W op.b fn.g fn.b
  // tri_in  18..33 : same layout
  // tas     34..43 : ln.g ln.b q.W k.W v.W bias.W out.W out.b gate.W gate.b
  // tae     44..53 : same layout
  // pt      54..59 : ln.g ln.b l1.W l1.b l2.W l2.b       (128->512->128)
  // attn    60..71 : ln_s.g ln_s.b ln_z.g ln_z.b q.W k.W v.W bias.W out.W out.b gate.W gate.b
  // st      72..77 : ln.g ln.b l1.W l1.b l2.W l2.b       (384->1536->384)
  auto prm = [&](int i) { return (const float*)d_in[i]; };
  const float* single_in = prm(0);
  const float* pair_in   = prm(1);

  float* singlebuf = (float*)d_out;                   // [L,CS]
  float* pairbuf   = (float*)d_out + (size_t)L * CS;  // [L,L,CZ]

  // --- workspace carve (bump allocator, 256B aligned) ---
  char* ws = (char*)d_ws;
  size_t off = 0;
  auto carve = [&](size_t bytes) {
    char* p = ws + off;
    off += (bytes + 255) & ~(size_t)255;
    return p;
  };
  float* zf = (float*)carve(NP * CZ * 4);
  float* t0 = (float*)carve(NP * CZ * 4);
  float* t1 = (float*)carve(NP * CZ * 4);
  float* t2 = (float*)carve(NP * CZ * 4);
  float* t3 = (float*)carve(NP * CZ * 4);
  char*  U  = carve(NP * 512 * 2);  // union region: hidden(bf16) OR packs+bias
  __bf16* bfA   = (__bf16*)U;                      // 8,388,608 elems
  __bf16* bfB   = bfA + (size_t)CZ * NP;
  __bf16* bfC   = bfB + (size_t)CZ * NP;
  float*  biasb = (float*)(bfC + (size_t)CZ * NP); // [NP, 8]
  __bf16* hid   = (__bf16*)U;                      // [NP, 512] (aliases packs)

  // --- launchers ---
  auto ln = [&](const float* x, const float* g, const float* b, float* y,
                size_t rows, int C) {
    ln_kernel<<<(unsigned)rows, 128, 0, stream>>>(x, g, b, y, C);
  };
  auto gemm_ff = [&](const float* A, const float* W, const float* b, float* Y,
                     int M, int N, int K, int relu) {
    dim3 g((N + 63) / 64, (M + 63) / 64);
    gemm_kernel<float, float><<<g, 128, 0, stream>>>(A, W, b, Y, M, N, K, relu);
  };
  auto gemm_fb = [&](const float* A, const float* W, const float* b, __bf16* Y,
                     int M, int N, int K, int relu) {
    dim3 g((N + 63) / 64, (M + 63) / 64);
    gemm_kernel<float, __bf16><<<g, 128, 0, stream>>>(A, W, b, Y, M, N, K, relu);
  };
  auto gemm_bf = [&](const __bf16* A, const float* W, const float* b, float* Y,
                     int M, int N, int K, int relu) {
    dim3 g((N + 63) / 64, (M + 63) / 64);
    gemm_kernel<__bf16, float><<<g, 128, 0, stream>>>(A, W, b, Y, M, N, K, relu);
  };
  auto eltgrid = [](size_t n) { return (unsigned)((n + 255) / 256); };
  auto sigmul = [&](const float* a, const float* g, float* y, size_t n) {
    sigmul_kernel<<<eltgrid(n), 256, 0, stream>>>(a, g, y, n);
  };
  auto gate_res = [&](const float* base, const float* gl, const float* x,
                      float* out, size_t n) {
    gate_residual_kernel<<<eltgrid(n), 256, 0, stream>>>(base, gl, x, out, n);
  };
  auto addi = [&](float* y, const float* x, size_t n) {
    add_kernel<<<eltgrid(n), 256, 0, stream>>>(y, x, n);
  };
  auto pack_cik = [&](const float* in, __bf16* out, int mode) {
    pack_cik_kernel<<<eltgrid((size_t)CZ * NP), 256, 0, stream>>>(in, out, mode);
  };
  auto pack_heads = [&](const float* in, __bf16* out) {
    pack_heads_kernel<<<eltgrid((size_t)H * NP * 16), 256, 0, stream>>>(in, out);
  };
  auto transpose = [&](const float* in, float* out) {
    transpose_kernel<<<eltgrid(NP * CZ), 256, 0, stream>>>(in, out);
  };

  // ---------------- triangle multiplicative update ----------------
  auto tri_mul = [&](const float* pairSrc, float* pairDst, int b, bool outgoing) {
    ln(pairSrc, prm(b + 0), prm(b + 1), zf, NP, CZ);
    gemm_ff(zf, prm(b + 2), prm(b + 3), t0, NP, CZ, CZ, 0);   // lp
    gemm_ff(zf, prm(b + 6), prm(b + 7), t1, NP, CZ, CZ, 0);   // lg
    sigmul(t0, t1, t0, NP * CZ);                              // left
    gemm_ff(zf, prm(b + 4), prm(b + 5), t1, NP, CZ, CZ, 0);   // rp
    gemm_ff(zf, prm(b + 8), prm(b + 9), t2, NP, CZ, CZ, 0);   // rg
    sigmul(t1, t2, t1, NP * CZ);                              // right
    pack_cik(t0, bfA, outgoing ? 0 : 1);
    pack_cik(t1, bfB, outgoing ? 0 : 1);
    tri_gemm_kernel<<<dim3(L / 64, L / 64, CZ), 128, 0, stream>>>(bfA, bfB, t2);
    ln(t2, prm(b + 14), prm(b + 15), zf, NP, CZ);             // fn
    gemm_ff(zf, prm(b + 12), prm(b + 13), t0, NP, CZ, CZ, 0); // op
    gemm_ff(pairSrc, prm(b + 10), prm(b + 11), t1, NP, CZ, CZ, 0); // og(pair)
    gate_res(pairSrc, t1, t0, pairDst, NP * CZ);
  };

  // ---------------- triangle attention ----------------
  auto tri_attn = [&](int b, bool ending) {
    float* pw = pairbuf;
    if (ending) { transpose(pairbuf, t3); pw = t3; }
    ln(pw, prm(b + 0), prm(b + 1), zf, NP, CZ);
    gemm_ff(zf, prm(b + 2), nullptr, t0, NP, CZ, CZ, 0); pack_heads(t0, bfA); // q
    gemm_ff(zf, prm(b + 3), nullptr, t0, NP, CZ, CZ, 0); pack_heads(t0, bfB); // k
    gemm_ff(zf, prm(b + 4), nullptr, t0, NP, CZ, CZ, 0); pack_heads(t0, bfC); // v
    gemm_ff(zf, prm(b + 5), nullptr, biasb, NP, H, CZ, 0);                    // bias
    tri_attn_kernel<<<dim3(L / 16, L, H), 32, 0, stream>>>(bfA, bfB, bfC,
                                                           biasb, t1);
    gemm_ff(t1, prm(b + 6), prm(b + 7), t0, NP, CZ, CZ, 0);   // out proj
    gemm_ff(pw, prm(b + 8), prm(b + 9), t2, NP, CZ, CZ, 0);   // gate(pair)
    gate_res(pw, t2, t0, pw, NP * CZ);
    if (ending) transpose(pw, pairbuf);
  };

  // ================= pipeline =================
  tri_mul(pair_in, pairbuf, 2, true);    // outgoing
  tri_mul(pairbuf, pairbuf, 18, false);  // incoming
  tri_attn(34, false);                   // starting node
  tri_attn(44, true);                    // ending node

  // pair transition (128 -> 512 -> 128, hidden in bf16)
  ln(pairbuf, prm(54), prm(55), zf, NP, CZ);
  gemm_fb(zf, prm(56), prm(57), hid, NP, 512, CZ, 1);
  gemm_bf(hid, prm(58), prm(59), t0, NP, CZ, 512, 0);
  addi(pairbuf, t0, NP * CZ);

  // single attention with pair bias
  {
    const size_t NS = (size_t)L * CS;
    float* sln   = t0;
    float* sq    = t0 + NS;
    float* sk    = t0 + 2 * NS;
    float* sv    = t0 + 3 * NS;
    float* soatt = t1;
    float* sout  = t1 + NS;
    float* sgate = t1 + 2 * NS;

    ln(single_in, prm(60), prm(61), sln, L, CS);
    ln(pairbuf, prm(62), prm(63), zf, NP, CZ);
    gemm_ff(zf, prm(67), nullptr, biasb, NP, H, CZ, 0);   // pair bias
    gemm_ff(sln, prm(64), nullptr, sq, L, CS, CS, 0);
    gemm_ff(sln, prm(65), nullptr, sk, L, CS, CS, 0);
    gemm_ff(sln, prm(66), nullptr, sv, L, CS, CS, 0);
    attn_single_kernel<<<dim3(L, H), 64, 0, stream>>>(sq, sk, sv, biasb, soatt);
    gemm_ff(soatt, prm(68), prm(69), sout, L, CS, CS, 0);
    gemm_ff(sln, prm(70), prm(71), sgate, L, CS, CS, 0);  // gate(LN(single))
    gate_res(single_in, sgate, sout, singlebuf, NS);

    // single transition (384 -> 1536 -> 384)
    ln(singlebuf, prm(72), prm(73), sln, L, CS);
    gemm_fb(sln, prm(74), prm(75), hid, L, 1536, CS, 1);
    gemm_bf(hid, prm(76), prm(77), sout, L, CS, 1536, 0);
    addi(singlebuf, sout, NS);
  }
}
